// Convattn_2765958938637
// MI455X (gfx1250) — matrix operand order repaired and checked by reading, not compiled, not run on hardware
//
#include <hip/hip_runtime.h>

// ---------------- problem constants ----------------
#define DIMC   512
#define NTT    256      // 16x16 temp tokens
#define NAA    484      // 22x22 target tokens
#define NTOK   740      // NTT + NAA
#define NPAD   768      // padded to 48 tiles of 16
#define HEADS  8
#define HD     64
#define BATCH  16
#define KTILES (NPAD/16)   // 48
#define QTILES 47          // ceil(740/16)

static constexpr float SM_SCALE = 0.125f;   // 64^-0.5
static constexpr float NEG_BIG  = -3.0e38f;

// ---------------- vector types for WMMA ----------------
typedef __attribute__((ext_vector_type(16))) __bf16 v16bf;
typedef __attribute__((ext_vector_type(8)))  __bf16 v8bf;
typedef __attribute__((ext_vector_type(8)))  float  v8f;

__device__ __forceinline__ v8f wmma_bf16(v16bf a, v16bf b, v8f c) {
    return __builtin_amdgcn_wmma_f32_16x16x32_bf16(
        /*neg_a=*/false, a, /*neg_b=*/false, b,
        /*c_mod=*/(short)0, c, /*reuse_a=*/false, /*reuse_b=*/false);
}

// build v16bf from two contiguous 8-element (16-byte) chunks
__device__ __forceinline__ v16bf pack16(const __bf16* lo_p, const __bf16* hi_p) {
    v8bf lo = *(const v8bf*)lo_p;
    v8bf hi = *(const v8bf*)hi_p;
    v16bf r;
#pragma unroll
    for (int i = 0; i < 8; ++i) { r[i] = lo[i]; r[8 + i] = hi[i]; }
    return r;
}

// =====================================================================
// Kernel 1: depthwise 3x3 conv + BN over the two token grids -> bf16
// tok layout: [B][NPAD][DIMC], rows >= NTOK zeroed.
// =====================================================================
__global__ __launch_bounds__(256)
void conv_bn_kernel(const float* __restrict__ x,      // [B][NTOK][DIMC]
                    const float* __restrict__ w,      // [DIMC][1][3][3]
                    const float* __restrict__ g,
                    const float* __restrict__ bb,
                    const float* __restrict__ mm,
                    const float* __restrict__ vv,
                    __bf16* __restrict__ tok)         // [B][NPAD][DIMC]
{
    const int n = blockIdx.x;
    const int b = blockIdx.y;
    for (int ch = threadIdx.x; ch < DIMC; ch += blockDim.x) {
        float y = 0.0f;
        if (n < NTOK) {
            int L, base, r, c;
            if (n < NTT) { L = 16; base = 0;   r = n >> 4;  c = n & 15; }
            else         { L = 22; base = NTT; int na = n - NTT; r = na / 22; c = na % 22; }
#pragma unroll
            for (int dr = -1; dr <= 1; ++dr) {
#pragma unroll
                for (int dc = -1; dc <= 1; ++dc) {
                    int rr = r + dr, cc = c + dc;
                    if (rr >= 0 && rr < L && cc >= 0 && cc < L) {
                        int np = base + rr * L + cc;
                        y += w[ch * 9 + (dr + 1) * 3 + (dc + 1)] *
                             x[((size_t)b * NTOK + np) * DIMC + ch];
                    }
                }
            }
            float inv = g[ch] * rsqrtf(vv[ch] + 1e-5f);
            y = y * inv + (bb[ch] - mm[ch] * inv);
        }
        tok[((size_t)b * NPAD + n) * DIMC + ch] = (__bf16)y;
    }
}

// =====================================================================
// Kernel 2: fp32 -> bf16 weight conversion
// =====================================================================
__global__ __launch_bounds__(256)
void cvt_bf16_kernel(const float* __restrict__ in, __bf16* __restrict__ out, int n)
{
    int i = blockIdx.x * blockDim.x + threadIdx.x;
    if (i < n) out[i] = (__bf16)in[i];
}

// =====================================================================
// Kernel 3: WMMA GEMM  out[b][n][d] = sum_c A[b][n][c] * W[d][c]
// 128 threads = 4 waves; each wave owns a 16(M) x 64(N) tile, fragments
// loaded straight from global (K-major => two b128 loads per fragment),
// no LDS, no barriers -> compiler can pipeline loads against WMMA.
// mode 0: bf16 out [B][NPAD][DIMC]
// mode 1: bf16 transposed out [B][DIMC][NPAD]
// mode 2: f32 out [B][NTOK][DIMC] + bias, rows clipped to NTOK
// =====================================================================
__global__ __launch_bounds__(128)
void gemm_tok_kernel(const __bf16* __restrict__ A,    // [B][NPAD][DIMC]
                     const __bf16* __restrict__ W,    // [DIMC][DIMC]
                     __bf16* __restrict__ outB,
                     __bf16* __restrict__ outT,
                     float*  __restrict__ outF,
                     const float* __restrict__ fbias,
                     int mode)
{
    const int b    = blockIdx.z;
    const int m0   = blockIdx.x * 16;
    const int tid  = threadIdx.x;
    const int lane = tid & 31;
    const int wave = tid >> 5;
    const int col  = lane & 15;
    const int half = lane >> 4;
    const int n0   = blockIdx.y * 256 + wave * 64;    // wave's 64 output cols

    v8f acc[4] = {{}, {}, {}, {}};

    const __bf16* arow = &A[((size_t)b * NPAD + m0 + col) * DIMC];

#pragma unroll 4
    for (int k0 = 0; k0 < DIMC; k0 += 32) {
        // A fragment: row M = col, chunks [half*8,+8) and [16+half*8,+8)
        v16bf afr = pack16(arow + k0 + half * 8,
                           arow + k0 + 16 + half * 8);
#pragma unroll
        for (int t = 0; t < 4; ++t) {
            // B fragment: output column = n0+t*16+col, K chunks [half*8,+8) and [16+half*8,+8)
            const __bf16* wr = &W[(size_t)(n0 + t * 16 + col) * DIMC + k0 + half * 8];
            v16bf bfr = pack16(wr, wr + 16);
            acc[t] = wmma_bf16(afr, bfr, acc[t]);
        }
    }

#pragma unroll
    for (int t = 0; t < 4; ++t) {
        const int dglob = n0 + t * 16 + col;
#pragma unroll
        for (int r = 0; r < 8; ++r) {
            int M   = r + half * 8;
            int row = m0 + M;
            float v = acc[t][r];
            if (mode == 0) {
                outB[((size_t)b * NPAD + row) * DIMC + dglob] = (__bf16)v;
            } else if (mode == 1) {
                outT[((size_t)b * DIMC + dglob) * NPAD + row] = (__bf16)v;
            } else if (row < NTOK) {
                outF[((size_t)b * NTOK + row) * DIMC + dglob] = v + fbias[dglob];
            }
        }
    }
}

// =====================================================================
// Kernel 4: assemble relative-position bias -> [H][NPAD][NPAD] fp32
// =====================================================================
__global__ __launch_bounds__(256)
void bias_kernel(const float* __restrict__ rpb_target, // [43*43][H]
                 const float* __restrict__ rpb_temp,   // [31*31][H]
                 const float* __restrict__ t2a_table,  // [H][NTT]
                 const float* __restrict__ a2t_table,  // [H][NAA]
                 const float* __restrict__ t2a_line,   // [H][NTT]
                 const float* __restrict__ a2t_line,   // [H][NAA]
                 float* __restrict__ biasb)            // [H][NPAD][NPAD]
{
    const int i = blockIdx.x;
    const int h = blockIdx.y;
    for (int j = threadIdx.x; j < NPAD; j += blockDim.x) {
        float v = 0.0f;
        if (i < NTOK && j < NTOK) {
            if (i < NTT && j < NTT) {
                int dy = (i >> 4) - (j >> 4) + 15;
                int dx = (i & 15) - (j & 15) + 15;
                v = rpb_temp[(dy * 31 + dx) * HEADS + h];
            } else if (i < NTT) {
                v = t2a_table[h * NTT + i] + a2t_line[h * NAA + (j - NTT)];
            } else if (j < NTT) {
                v = a2t_table[h * NAA + (i - NTT)] + t2a_line[h * NTT + j];
            } else {
                int ia = i - NTT, ja = j - NTT;
                int dy = ia / 22 - ja / 22 + 21;
                int dx = ia % 22 - ja % 22 + 21;
                v = rpb_target[(dy * 43 + dx) * HEADS + h];
            }
        }
        biasb[((size_t)h * NPAD + i) * NPAD + j] = v;
    }
}

// =====================================================================
// Kernel 5: fused attention for one (b, h, 16-query tile)
// 128 threads = 4 waves. S row-block kept in LDS fp32.
// =====================================================================
__global__ __launch_bounds__(128)
void attn_kernel(const __bf16* __restrict__ qp,   // [B][NPAD][DIMC]
                 const __bf16* __restrict__ kp,   // [B][NPAD][DIMC]
                 const __bf16* __restrict__ vpT,  // [B][DIMC][NPAD]
                 const float*  __restrict__ biasb,// [H][NPAD][NPAD]
                 __bf16* __restrict__ ao)         // [B][NPAD][DIMC]
{
    __shared__ __align__(16) float S[16 * NPAD];   // 48 KB
    __shared__ float redm[16 * 8];
    __shared__ float reds[16 * 8];

    const int qt = blockIdx.x;        // 0..QTILES-1
    const int h  = blockIdx.y;
    const int b  = blockIdx.z;
    const int q0 = qt * 16;
    const int hbase = h * HD;

    const int tid  = threadIdx.x;
    const int lane = tid & 31;
    const int wave = tid >> 5;
    const int col  = lane & 15;
    const int half = lane >> 4;

    // ---- Q fragments (A operand), K = 64 split into two 32-chunks ----
    const __bf16* qrow = &qp[((size_t)b * NPAD + q0 + col) * DIMC + hbase];
    v16bf a0 = pack16(qrow + half * 8,        qrow + 16 + half * 8);
    v16bf a1 = pack16(qrow + 32 + half * 8,   qrow + 48 + half * 8);

    // ---- scores: key tiles strided across the 4 waves ----
    for (int kt = wave; kt < KTILES; kt += 4) {
        const __bf16* krow = &kp[((size_t)b * NPAD + kt * 16 + col) * DIMC + hbase];
        v16bf b0 = pack16(krow + half * 8,      krow + 16 + half * 8);
        v16bf b1 = pack16(krow + 32 + half * 8, krow + 48 + half * 8);
        v8f acc = {};
        acc = wmma_bf16(a0, b0, acc);
        acc = wmma_bf16(a1, b1, acc);
        int key = kt * 16 + col;
#pragma unroll
        for (int r = 0; r < 8; ++r) {
            int M = r + half * 8;
            float s = acc[r] * SM_SCALE +
                      biasb[((size_t)h * NPAD + q0 + M) * NPAD + key];
            if (key >= NTOK) s = NEG_BIG;
            S[M * NPAD + key] = s;
        }
    }
    __syncthreads();

    // ---- softmax over each of the 16 rows (8 threads per row) ----
    const int row  = tid & 15;
    const int slot = tid >> 4;     // 0..7
    float mx = NEG_BIG;
    for (int j = slot; j < NPAD; j += 8) mx = fmaxf(mx, S[row * NPAD + j]);
    redm[row * 8 + slot] = mx;
    __syncthreads();
    mx = redm[row * 8];
#pragma unroll
    for (int s = 1; s < 8; ++s) mx = fmaxf(mx, redm[row * 8 + s]);
    float sum = 0.0f;
    for (int j = slot; j < NPAD; j += 8) {
        float e = __expf(S[row * NPAD + j] - mx);
        S[row * NPAD + j] = e;           // unnormalized P, fp32
        sum += e;
    }
    reds[row * 8 + slot] = sum;
    __syncthreads();
    if (slot == 0) {
        float t = 0.0f;
#pragma unroll
        for (int s = 0; s < 8; ++s) t += reds[row * 8 + s];
        redm[row * 8] = 1.0f / t;        // stash inverse row-sum
    }
    __syncthreads();

    // ---- O = P * V : each wave owns 16 of the 64 head channels ----
    const int d0 = wave * 16;
    v8f oacc = {};
    for (int c0 = 0; c0 < NPAD; c0 += 32) {
        // A fragment from fp32 P in LDS, converted to bf16 at gather time
        const float* srow = &S[col * NPAD + c0];
        v8f lo = *(const v8f*)&srow[half * 8];
        v8f hi = *(const v8f*)&srow[16 + half * 8];
        v16bf pa;
#pragma unroll
        for (int i = 0; i < 8; ++i) { pa[i] = (__bf16)lo[i]; pa[8 + i] = (__bf16)hi[i]; }
        // B fragment: vpT[channel][key] is contiguous along K
        const __bf16* vrow =
            &vpT[((size_t)b * DIMC + hbase + d0 + col) * NPAD + c0 + half * 8];
        v16bf vb = pack16(vrow, vrow + 16);
        oacc = wmma_bf16(pa, vb, oacc);
    }
#pragma unroll
    for (int r = 0; r < 8; ++r) {
        int M = r + half * 8;
        float o = oacc[r] * redm[M * 8];
        ao[((size_t)b * NPAD + q0 + M) * DIMC + hbase + d0 + col] = (__bf16)o;
    }
}

// =====================================================================
// Host launcher
// =====================================================================
extern "C" void kernel_launch(void* const* d_in, const int* in_sizes, int n_in,
                              void* d_out, int out_size, void* d_ws, size_t ws_size,
                              hipStream_t stream)
{
    (void)in_sizes; (void)n_in; (void)out_size; (void)ws_size;

    const float* x     = (const float*)d_in[0];
    const float* wcq   = (const float*)d_in[1];
    const float* bnq_g = (const float*)d_in[2];
    const float* bnq_b = (const float*)d_in[3];
    const float* bnq_m = (const float*)d_in[4];
    const float* bnq_v = (const float*)d_in[5];
    const float* wq    = (const float*)d_in[6];
    const float* wck   = (const float*)d_in[7];
    const float* bnk_g = (const float*)d_in[8];
    const float* bnk_b = (const float*)d_in[9];
    const float* bnk_m = (const float*)d_in[10];
    const float* bnk_v = (const float*)d_in[11];
    const float* wk    = (const float*)d_in[12];
    const float* wcv   = (const float*)d_in[13];
    const float* bnv_g = (const float*)d_in[14];
    const float* bnv_b = (const float*)d_in[15];
    const float* bnv_m = (const float*)d_in[16];
    const float* bnv_v = (const float*)d_in[17];
    const float* wv    = (const float*)d_in[18];
    const float* wproj = (const float*)d_in[19];
    const float* bproj = (const float*)d_in[20];
    const float* rpb_target = (const float*)d_in[21];
    const float* rpb_temp   = (const float*)d_in[22];
    const float* t2a_table  = (const float*)d_in[23];
    const float* a2t_table  = (const float*)d_in[24];
    const float* t2a_line   = (const float*)d_in[25];
    const float* a2t_line   = (const float*)d_in[26];

    // ---- workspace layout ----
    const size_t ST  = (size_t)BATCH * NPAD * DIMC * 2;  // bf16 token buffer
    const size_t WSZ = (size_t)DIMC * DIMC * 2;          // bf16 weight
    char* ws = (char*)d_ws;
    __bf16* qtok = (__bf16*)(ws + 0 * ST);
    __bf16* ktok = (__bf16*)(ws + 1 * ST);
    __bf16* vtok = (__bf16*)(ws + 2 * ST);
    __bf16* qp   = (__bf16*)(ws + 3 * ST);
    __bf16* kp   = (__bf16*)(ws + 4 * ST);
    __bf16* vpT  = (__bf16*)(ws + 5 * ST);               // [B][DIMC][NPAD]
    __bf16* aob  = (__bf16*)(ws + 6 * ST);
    __bf16* wq_bf    = (__bf16*)(ws + 7 * ST + 0 * WSZ);
    __bf16* wk_bf    = (__bf16*)(ws + 7 * ST + 1 * WSZ);
    __bf16* wv_bf    = (__bf16*)(ws + 7 * ST + 2 * WSZ);
    __bf16* wproj_bf = (__bf16*)(ws + 7 * ST + 3 * WSZ);
    float*  biasb    = (float*) (ws + 7 * ST + 4 * WSZ); // [H][NPAD][NPAD] f32

    // 1) conv + BN -> bf16 tokens (zero-padded to NPAD rows)
    dim3 cgrid(NPAD, BATCH);
    conv_bn_kernel<<<cgrid, 256, 0, stream>>>(x, wcq, bnq_g, bnq_b, bnq_m, bnq_v, qtok);
    conv_bn_kernel<<<cgrid, 256, 0, stream>>>(x, wck, bnk_g, bnk_b, bnk_m, bnk_v, ktok);
    conv_bn_kernel<<<cgrid, 256, 0, stream>>>(x, wcv, bnv_g, bnv_b, bnv_m, bnv_v, vtok);

    // 2) weight conversion
    const int wn = DIMC * DIMC;
    cvt_bf16_kernel<<<wn / 256, 256, 0, stream>>>(wq,    wq_bf,    wn);
    cvt_bf16_kernel<<<wn / 256, 256, 0, stream>>>(wk,    wk_bf,    wn);
    cvt_bf16_kernel<<<wn / 256, 256, 0, stream>>>(wv,    wv_bf,    wn);
    cvt_bf16_kernel<<<wn / 256, 256, 0, stream>>>(wproj, wproj_bf, wn);

    // 3) q/k/v projections via WMMA GEMM (4 waves x 16x64 tiles, LDS-free)
    dim3 ggrid(NPAD / 16, DIMC / 256, BATCH);
    gemm_tok_kernel<<<ggrid, 128, 0, stream>>>(qtok, wq_bf, qp, nullptr, nullptr, nullptr, 0);
    gemm_tok_kernel<<<ggrid, 128, 0, stream>>>(ktok, wk_bf, kp, nullptr, nullptr, nullptr, 0);
    gemm_tok_kernel<<<ggrid, 128, 0, stream>>>(vtok, wv_bf, nullptr, vpT, nullptr, nullptr, 1);

    // 4) relative-position bias table
    dim3 bgrid(NPAD, HEADS);
    bias_kernel<<<bgrid, 256, 0, stream>>>(rpb_target, rpb_temp, t2a_table, a2t_table,
                                           t2a_line, a2t_line, biasb);

    // 5) fused attention (scores + softmax + PV) via WMMA
    dim3 agrid(QTILES, HEADS, BATCH);
    attn_kernel<<<agrid, 128, 0, stream>>>(qp, kp, vpT, biasb, aob);

    // 6) output projection + bias -> fp32 d_out [B][NTOK][DIMC]
    gemm_tok_kernel<<<ggrid, 128, 0, stream>>>(aob, wproj_bf, nullptr, nullptr,
                                               (float*)d_out, bproj, 2);
}
